// TemporalAttention_11656541241926
// MI455X (gfx1250) — compile-verified, matrix operand order checked by reading, not run on hardware
//
#include <hip/hip_runtime.h>
#include <hip/hip_bf16.h>

// ---------------------------------------------------------------------------
// Windowed causal attention on gfx1250 via v_wmma_f32_16x16x32_f16.
//   x:[16,4096,512] -> 512 windows of 128 tokens, 8 heads x hd=64.
// Pipeline: cvt weights -> QKV gemm -> per-(win,head) attention -> proj gemm.
// QKV gemm: A (f32->f16 fused) and B both staged in double-buffered LDS.
// Proj gemm: B staged in LDS, f16 A register-prefetched.
// ---------------------------------------------------------------------------

typedef __attribute__((ext_vector_type(16))) _Float16 v16h;
typedef __attribute__((ext_vector_type(8)))  _Float16 v8h;
typedef __attribute__((ext_vector_type(8)))  float    v8f;

union F16Frag { v16h v; _Float16 h[16]; v8h q2[2]; };
union F32Acc  { v8f  v; float f[8]; };
union H32     { v8h v[4]; _Float16 h[32]; };

#define TA_DIM   512
#define TA_HEADS 8
#define TA_WIN   128
#define TA_HD    64
#define TA_ROWS  65536              // 16*4096 tokens
#define TA_NWIN  512                // TA_ROWS / TA_WIN
#define TA_ASTRIDE 72               // padded LDS row stride (halves)
#define TA_BSTRIDE 72

// ---- fragment loaders -----------------------------------------------------
// 16-bit A/B layout (CDNA5 ISA 7.12.2): lane holds row/col = lane&15,
// K offset base = (lane&16)?8:0, elements 0..7 -> K+0..7, 8..15 -> K+16..23.
// So a fragment is two contiguous 16-byte loads at +0 and +16 halves.

static __device__ inline v16h load_frag_h(const _Float16* p) {
  F16Frag u;
  u.q2[0] = *(const v8h*)(p);
  u.q2[1] = *(const v8h*)(p + 16);
  return u.v;
}

// ---- weight conversion ----------------------------------------------------
__global__ void ta_cvt_f16(const float* __restrict__ in,
                           _Float16* __restrict__ out, int n) {
  int i = blockIdx.x * 256 + threadIdx.x;
  if (i < n) out[i] = (_Float16)in[i];
}

// ---- QKV projection: [65536,512]f32 x [512,1536] -> q/k/vT f16 ------------
// grid (24, 512), block 256 (8 waves). Block: 128 rows x 64 cols.
// A tile (128x64, f32->f16 in staging) and B tile (64x64) double-buffered
// in LDS; global loads issued at loop top, committed after the WMMAs.
__global__ __launch_bounds__(256) void ta_gemm_qkv(
    const float* __restrict__ x,       // [65536,512]
    const _Float16* __restrict__ wq,   // [1536,512] (row n holds K contiguous)
    _Float16* __restrict__ qb,         // [512][8][128][64]
    _Float16* __restrict__ kb,         // [512][8][128][64]
    _Float16* __restrict__ vtb) {      // [512][8][64][128]
  __shared__ __align__(16) _Float16 As[2][128 * TA_ASTRIDE];  // 2 x 18 KB
  __shared__ __align__(16) _Float16 Bs[2][64 * TA_BSTRIDE];   // 2 x  9 KB

  const int tid  = threadIdx.x;
  const int lane = tid & 31;
  const int wv   = tid >> 5;
  const int ln16 = lane & 15;
  const int koff = (lane & 16) ? 8 : 0;
  const int hi8  = (lane & 16) ? 8 : 0;
  const int win  = blockIdx.y;
  const int n0   = blockIdx.x * 64;

  // cooperative A stager: thread -> (row arow, 32-half slice aks)
  const int arow = tid >> 1;
  const int aks  = (tid & 1) * 32;
  const float* asrc = x + ((size_t)win * 128 + arow) * TA_DIM + aks;
  // cooperative B stager: thread -> (row bn, 16-half slice bk)
  const int bn = tid & 63;
  const int bk = (tid >> 6) * 16;
  const _Float16* bsrc = wq + (size_t)(n0 + bn) * TA_DIM + bk;

  F32Acc acc[4];
#pragma unroll
  for (int t = 0; t < 4; ++t)
#pragma unroll
    for (int e = 0; e < 8; ++e) acc[t].f[e] = 0.0f;

  // prologue: stage chunk 0
  {
    float t[32];
#pragma unroll
    for (int j = 0; j < 8; ++j)
      *(float4*)&t[j * 4] = *(const float4*)(asrc + j * 4);
    v8h b0 = *(const v8h*)(bsrc);
    v8h b1 = *(const v8h*)(bsrc + 8);
    H32 u;
#pragma unroll
    for (int e = 0; e < 32; ++e) u.h[e] = (_Float16)t[e];
#pragma unroll
    for (int j = 0; j < 4; ++j)
      *(v8h*)(&As[0][arow * TA_ASTRIDE + aks + j * 8]) = u.v[j];
    *(v8h*)(&Bs[0][bn * TA_BSTRIDE + bk])     = b0;
    *(v8h*)(&Bs[0][bn * TA_BSTRIDE + bk + 8]) = b1;
  }
  __syncthreads();

  for (int c = 0; c < 8; ++c) {            // 8 chunks x 64 K
    float apre[32];
    v8h bp0, bp1;
    if (c < 7) {                            // issue chunk c+1 global loads
      const float* as = asrc + (c + 1) * 64;
#pragma unroll
      for (int j = 0; j < 8; ++j)
        *(float4*)&apre[j * 4] = *(const float4*)(as + j * 4);
      const _Float16* bs = bsrc + (c + 1) * 64;
      bp0 = *(const v8h*)(bs);
      bp1 = *(const v8h*)(bs + 8);
    }
    // compute chunk c: two 32-K substeps x 4 col-tiles, A and B from LDS
    const _Float16* asl = &As[c & 1][(wv * 16 + ln16) * TA_ASTRIDE];
    const _Float16* bsl = &Bs[c & 1][0];
#pragma unroll
    for (int sub = 0; sub < 2; ++sub) {
      v16h a = load_frag_h(asl + sub * 32 + koff);
#pragma unroll
      for (int nt = 0; nt < 4; ++nt) {
        const int nl = nt * 16 + ln16;
        v16h b = load_frag_h(bsl + nl * TA_BSTRIDE + sub * 32 + koff);
        acc[nt].v = __builtin_amdgcn_wmma_f32_16x16x32_f16(
            false, a, false, b, (short)0, acc[nt].v, false, false);
      }
    }
    if (c < 7) {                            // convert + commit chunk c+1
      H32 u;
#pragma unroll
      for (int e = 0; e < 32; ++e) u.h[e] = (_Float16)apre[e];
      _Float16* ad = &As[(c + 1) & 1][arow * TA_ASTRIDE + aks];
#pragma unroll
      for (int j = 0; j < 4; ++j) *(v8h*)(ad + j * 8) = u.v[j];
      _Float16* bd = &Bs[(c + 1) & 1][bn * TA_BSTRIDE + bk];
      *(v8h*)(bd)     = bp0;
      *(v8h*)(bd + 8) = bp1;
    }
    __syncthreads();
  }

#pragma unroll
  for (int nt = 0; nt < 4; ++nt) {
    const int n    = n0 + nt * 16 + ln16;   // 0..1535
    const int sec  = n >> 9;                // 0=q 1=k 2=v
    const int c    = n & 511;
    const int head = c >> 6;
    const int d    = c & 63;
    const size_t bh = (size_t)win * TA_HEADS + head;
#pragma unroll
    for (int r = 0; r < 8; ++r) {
      const int pos = wv * 16 + r + hi8;
      const _Float16 val = (_Float16)acc[nt].f[r];
      if (sec == 0)      qb[(bh * TA_WIN + pos) * TA_HD + d] = val;
      else if (sec == 1) kb[(bh * TA_WIN + pos) * TA_HD + d] = val;
      else               vtb[(bh * TA_HD + d) * TA_WIN + pos] = val;
    }
  }
}

// ---- attention: one block per (window, head) ------------------------------
// 8 waves; wave w -> query rows 16w..16w+15. S=QK^T (8 j-tiles x 2 ksteps),
// causal mask + softmax on f32 accumulators, P re-striped via LDS, PV.
__global__ __launch_bounds__(256) void ta_attn(
    const _Float16* __restrict__ qb,
    const _Float16* __restrict__ kb,
    const _Float16* __restrict__ vtb,
    _Float16* __restrict__ ao) {       // [512][128][512] merged heads
  __shared__ _Float16 Plds[8 * 16 * TA_WIN];   // 32 KB

  const int lane = threadIdx.x & 31;
  const int wv   = threadIdx.x >> 5;
  const int ln16 = lane & 15;
  const int koff = (lane & 16) ? 8 : 0;
  const int hi8  = (lane & 16) ? 8 : 0;
  const int win  = blockIdx.x >> 3;
  const int head = blockIdx.x & 7;
  const size_t bh = (size_t)win * TA_HEADS + head;
  const int i0 = wv * 16;

  // Q fragments for this wave's 16 rows (K = hd in two 32-steps)
  const _Float16* qrow = qb + (bh * TA_WIN + i0 + ln16) * TA_HD;
  v16h aq0 = load_frag_h(qrow + 0  + koff);
  v16h aq1 = load_frag_h(qrow + 32 + koff);

  F32Acc s[8];
#pragma unroll
  for (int jt = 0; jt < 8; ++jt)
#pragma unroll
    for (int e = 0; e < 8; ++e) s[jt].f[e] = 0.0f;

#pragma unroll
  for (int jt = 0; jt < 8; ++jt) {
    const _Float16* krow = kb + (bh * TA_WIN + jt * 16 + ln16) * TA_HD;
    v16h b0 = load_frag_h(krow + 0  + koff);
    v16h b1 = load_frag_h(krow + 32 + koff);
    s[jt].v = __builtin_amdgcn_wmma_f32_16x16x32_f16(
        false, aq0, false, b0, (short)0, s[jt].v, false, false);
    s[jt].v = __builtin_amdgcn_wmma_f32_16x16x32_f16(
        false, aq1, false, b1, (short)0, s[jt].v, false, false);
  }

  // causal mask + scale (hd^-0.5 = 0.125)
  const float scale = 0.125f;
#pragma unroll
  for (int jt = 0; jt < 8; ++jt)
#pragma unroll
    for (int r = 0; r < 8; ++r) {
      const int i = i0 + r + hi8;
      const int j = jt * 16 + ln16;
      s[jt].f[r] = (j <= i) ? s[jt].f[r] * scale : -3.0e38f;
    }

  // row softmax: row r lives in element r across one 16-lane half,
  // so shfl_xor 1/2/4/8 reduces over columns without mixing rows.
#pragma unroll
  for (int r = 0; r < 8; ++r) {
    float mx = -3.0e38f;
#pragma unroll
    for (int jt = 0; jt < 8; ++jt) mx = fmaxf(mx, s[jt].f[r]);
#pragma unroll
    for (int m = 1; m < 16; m <<= 1) mx = fmaxf(mx, __shfl_xor(mx, m, 32));
    float sum = 0.0f;
#pragma unroll
    for (int jt = 0; jt < 8; ++jt) {
      float p = __expf(s[jt].f[r] - mx);
      s[jt].f[r] = p;
      sum += p;
    }
#pragma unroll
    for (int m = 1; m < 16; m <<= 1) sum += __shfl_xor(sum, m, 32);
    const float inv = __builtin_amdgcn_rcpf(sum);
#pragma unroll
    for (int jt = 0; jt < 8; ++jt) s[jt].f[r] *= inv;
  }

  // C/D-layout -> A-layout re-striping through LDS
#pragma unroll
  for (int jt = 0; jt < 8; ++jt)
#pragma unroll
    for (int r = 0; r < 8; ++r)
      Plds[wv * (16 * TA_WIN) + (r + hi8) * TA_WIN + jt * 16 + ln16] =
          (_Float16)s[jt].f[r];
  __syncthreads();

  // O = P @ V : K = 128 in four 32-steps, N = hd in four 16-tiles
  F32Acc o[4];
#pragma unroll
  for (int t = 0; t < 4; ++t)
#pragma unroll
    for (int e = 0; e < 8; ++e) o[t].f[e] = 0.0f;

#pragma unroll
  for (int kt = 0; kt < 4; ++kt) {
    v16h ap = load_frag_h(&Plds[wv * (16 * TA_WIN) + ln16 * TA_WIN +
                                kt * 32 + koff]);
#pragma unroll
    for (int dt = 0; dt < 4; ++dt) {
      v16h bv = load_frag_h(vtb + (bh * TA_HD + dt * 16 + ln16) * TA_WIN +
                            kt * 32 + koff);
      o[dt].v = __builtin_amdgcn_wmma_f32_16x16x32_f16(
          false, ap, false, bv, (short)0, o[dt].v, false, false);
    }
  }

#pragma unroll
  for (int dt = 0; dt < 4; ++dt) {
    const int c = head * TA_HD + dt * 16 + ln16;
#pragma unroll
    for (int r = 0; r < 8; ++r) {
      const int i = i0 + r + hi8;
      ao[((size_t)win * TA_WIN + i) * TA_DIM + c] = (_Float16)o[dt].f[r];
    }
  }
}

// ---- output projection + bias: [65536,512]f16 x [512,512] -> f32 ----------
// B staged in LDS (double buffered), f16 A register-prefetched.
__global__ __launch_bounds__(256) void ta_gemm_proj(
    const _Float16* __restrict__ ao,   // [65536,512]
    const _Float16* __restrict__ wp,   // [512,512] (row n holds K contiguous)
    const float* __restrict__ bias,    // [512]
    float* __restrict__ out) {         // [65536,512]
  __shared__ __align__(16) _Float16 Bs[2][64 * TA_BSTRIDE];

  const int tid  = threadIdx.x;
  const int lane = tid & 31;
  const int wv   = tid >> 5;
  const int ln16 = lane & 15;
  const int koff = (lane & 16) ? 8 : 0;
  const int hi8  = (lane & 16) ? 8 : 0;
  const int m0   = blockIdx.y * 128 + wv * 16;
  const int n0   = blockIdx.x * 64;
  const _Float16* aptr = ao + (size_t)(m0 + ln16) * TA_DIM;

  const int bn = tid & 63;
  const int bk = (tid >> 6) * 16;
  const _Float16* bsrc = wp + (size_t)(n0 + bn) * TA_DIM + bk;

  F32Acc acc[4];
#pragma unroll
  for (int t = 0; t < 4; ++t)
#pragma unroll
    for (int e = 0; e < 8; ++e) acc[t].f[e] = 0.0f;

  {
    v8h b0 = *(const v8h*)(bsrc);
    v8h b1 = *(const v8h*)(bsrc + 8);
    *(v8h*)(&Bs[0][bn * TA_BSTRIDE + bk])     = b0;
    *(v8h*)(&Bs[0][bn * TA_BSTRIDE + bk + 8]) = b1;
  }
  v16h a0 = load_frag_h(aptr + 0 + koff);
  v16h a1 = load_frag_h(aptr + 32 + koff);
  __syncthreads();

  for (int c = 0; c < 8; ++c) {
    v8h bp0, bp1;
    v16h an0, an1;
    if (c < 7) {
      const _Float16* bs = bsrc + (c + 1) * 64;
      bp0 = *(const v8h*)(bs);
      bp1 = *(const v8h*)(bs + 8);
      an0 = load_frag_h(aptr + (c + 1) * 64 + 0 + koff);
      an1 = load_frag_h(aptr + (c + 1) * 64 + 32 + koff);
    }
    const _Float16* bsl = &Bs[c & 1][0];
#pragma unroll
    for (int nt = 0; nt < 4; ++nt) {
      const int nl = nt * 16 + ln16;
      v16h b = load_frag_h(bsl + nl * TA_BSTRIDE + 0 + koff);
      acc[nt].v = __builtin_amdgcn_wmma_f32_16x16x32_f16(
          false, a0, false, b, (short)0, acc[nt].v, false, false);
    }
#pragma unroll
    for (int nt = 0; nt < 4; ++nt) {
      const int nl = nt * 16 + ln16;
      v16h b = load_frag_h(bsl + nl * TA_BSTRIDE + 32 + koff);
      acc[nt].v = __builtin_amdgcn_wmma_f32_16x16x32_f16(
          false, a1, false, b, (short)0, acc[nt].v, false, false);
    }
    if (c < 7) {
      _Float16* bd = &Bs[(c + 1) & 1][bn * TA_BSTRIDE + bk];
      *(v8h*)(bd)     = bp0;
      *(v8h*)(bd + 8) = bp1;
      a0 = an0;
      a1 = an1;
    }
    __syncthreads();
  }

#pragma unroll
  for (int nt = 0; nt < 4; ++nt) {
    const int col = n0 + nt * 16 + ln16;
    const float bv = bias[col];
#pragma unroll
    for (int r = 0; r < 8; ++r) {
      const int row = m0 + r + hi8;
      out[(size_t)row * TA_DIM + col] = acc[nt].f[r] + bv;
    }
  }
}

// ---------------------------------------------------------------------------
extern "C" void kernel_launch(void* const* d_in, const int* in_sizes, int n_in,
                              void* d_out, int out_size, void* d_ws,
                              size_t ws_size, hipStream_t stream) {
  const float* x      = (const float*)d_in[0];
  const float* w_qkv  = (const float*)d_in[1];
  const float* w_proj = (const float*)d_in[2];
  const float* b_proj = (const float*)d_in[3];
  float* out = (float*)d_out;

  _Float16* ws = (_Float16*)d_ws;
  size_t off = 0;
  _Float16* wq16 = ws + off; off += (size_t)3 * TA_DIM * TA_DIM;      // 1536x512
  _Float16* wp16 = ws + off; off += (size_t)TA_DIM * TA_DIM;          // 512x512
  _Float16* qb   = ws + off; off += (size_t)TA_ROWS * TA_DIM;         // 64 MB
  _Float16* kb   = ws + off; off += (size_t)TA_ROWS * TA_DIM;
  _Float16* vtb  = ws + off; off += (size_t)TA_ROWS * TA_DIM;
  _Float16* ao   = ws + off; off += (size_t)TA_ROWS * TA_DIM;
  (void)ws_size; (void)in_sizes; (void)n_in; (void)out_size;

  const int nwq = 3 * TA_DIM * TA_DIM;
  const int nwp = TA_DIM * TA_DIM;
  ta_cvt_f16<<<(nwq + 255) / 256, 256, 0, stream>>>(w_qkv, wq16, nwq);
  ta_cvt_f16<<<(nwp + 255) / 256, 256, 0, stream>>>(w_proj, wp16, nwp);

  ta_gemm_qkv<<<dim3(24, TA_NWIN), 256, 0, stream>>>(x, wq16, qb, kb, vtb);
  ta_attn<<<TA_NWIN * TA_HEADS, 256, 0, stream>>>(qb, kb, vtb, ao);
  ta_gemm_proj<<<dim3(8, TA_ROWS / 128), 256, 0, stream>>>(ao, wp16, b_proj,
                                                           out);
}
